// PointNet_61546881352044
// MI455X (gfx1250) — matrix operand order repaired and checked by reading, not compiled, not run on hardware
//
#include <hip/hip_runtime.h>
#include <stdint.h>

typedef __attribute__((ext_vector_type(16))) _Float16 v16h;
typedef __attribute__((ext_vector_type(8)))  _Float16 v8h;
typedef __attribute__((ext_vector_type(8)))  float    v8f;

#define TPB 256

// ======================= elementwise / utility kernels =======================

__global__ void fill_f32_kernel(float* p, float v, long long n) {
  long long i = (long long)blockIdx.x * blockDim.x + threadIdx.x;
  if (i < n) p[i] = v;
}

__global__ void fill_u32_kernel(unsigned* p, unsigned v, long long n) {
  long long i = (long long)blockIdx.x * blockDim.x + threadIdx.x;
  if (i < n) p[i] = v;
}

__global__ void relu_kernel(float* p, long long n) {
  long long i = (long long)blockIdx.x * blockDim.x + threadIdx.x;
  if (i < n) p[i] = fmaxf(p[i], 0.f);
}

// decode order-preserving u32 max encoding back to float, in place
__global__ void decode_max_kernel(unsigned* p, long long n) {
  long long i = (long long)blockIdx.x * blockDim.x + threadIdx.x;
  if (i < n) {
    unsigned u = p[i];
    float f = (u & 0x80000000u) ? __uint_as_float(u ^ 0x80000000u)
                                : __uint_as_float(~u);
    ((float*)p)[i] = f;
  }
}

__global__ void reduce_sum_kernel(const float* __restrict__ w, long long n, float* out) {
  __shared__ float sh[TPB];
  long long i = (long long)blockIdx.x * blockDim.x + threadIdx.x;
  long long stride = (long long)gridDim.x * blockDim.x;
  float s = 0.f;
  for (; i < n; i += stride) s += w[i];
  sh[threadIdx.x] = s;
  __syncthreads();
  for (int k = TPB / 2; k > 0; k >>= 1) {
    if ((int)threadIdx.x < k) sh[threadIdx.x] += sh[threadIdx.x + k];
    __syncthreads();
  }
  if (threadIdx.x == 0) atomicAdd(out, sh[0]);
}

// Weight prep: W (Cin x Cout, f32, row-major) -> Wt (CoutPad x CinPad, f16, zero-padded)
__global__ void prep_weight_kernel(const float* __restrict__ W, _Float16* __restrict__ Wt,
                                   int Cin, int Cout, int CinPad, long long total) {
  long long e = (long long)blockIdx.x * blockDim.x + threadIdx.x;
  if (e >= total) return;
  int n = (int)(e / CinPad);
  int k = (int)(e % CinPad);
  Wt[e] = (n < Cout && k < Cin) ? (_Float16)W[(size_t)k * Cout + n] : (_Float16)0.f;
}

// ======================= FPS (one block per batch) =======================

__global__ void fps_kernel(const float* __restrict__ pos, int* __restrict__ idx,
                           int n, int m) {
  int b = blockIdx.x;
  const float* P = pos + (size_t)b * n * 3;
  int* I = idx + (size_t)b * m;
  __shared__ float sv[TPB];
  __shared__ int   si[TPB];
  __shared__ float lastp[3];
  __shared__ int   seli;
  int t = threadIdx.x;
  float mind[16];
  int cnt = (n + TPB - 1) / TPB;   // <= 16 for n <= 4096
#pragma unroll
  for (int j = 0; j < 16; ++j) mind[j] = 3.4e38f;
  if (t == 0) { I[0] = 0; seli = 0; }
  __syncthreads();
  for (int i = 1; i < m; ++i) {
    if (t == 0) {
      int s = seli;
      lastp[0] = P[s * 3 + 0]; lastp[1] = P[s * 3 + 1]; lastp[2] = P[s * 3 + 2];
    }
    __syncthreads();
    float lx = lastp[0], ly = lastp[1], lz = lastp[2];
    float bestv = -1.f; int besti = 0;
    for (int j = 0; j < cnt; ++j) {
      int p = t + j * TPB;
      if (p < n) {
        float dx = P[p * 3 + 0] - lx;
        float dy = P[p * 3 + 1] - ly;
        float dz = P[p * 3 + 2] - lz;
        float d = dx * dx + dy * dy + dz * dz;
        float mm = fminf(mind[j], d);
        mind[j] = mm;
        if (mm > bestv) { bestv = mm; besti = p; }
      }
    }
    sv[t] = bestv; si[t] = besti;
    __syncthreads();
    for (int s = TPB / 2; s > 0; s >>= 1) {
      if (t < s) {
        float v2 = sv[t + s]; int i2 = si[t + s];
        if (v2 > sv[t] || (v2 == sv[t] && i2 < si[t])) { sv[t] = v2; si[t] = i2; }
      }
      __syncthreads();
    }
    if (t == 0) { I[i] = si[0]; seli = si[0]; }
    __syncthreads();
  }
}

// ======================= gather rows along axis 1 =======================

__global__ void gather_rows_kernel(const float* __restrict__ src, const int* __restrict__ idx,
                                   float* __restrict__ dst, int n, int m, int C, long long total) {
  long long e = (long long)blockIdx.x * blockDim.x + threadIdx.x;
  if (e >= total) return;
  long long r = e / C; int c = (int)(e % C);     // r = b*m + q
  int b = (int)(r / m);
  dst[e] = src[((size_t)b * n + idx[r]) * C + c];
}

// ======================= brute-force KNN (thread per query) =======================

__global__ void knn_kernel(const float* __restrict__ pos, const float* __restrict__ qpos,
                           int* __restrict__ nidx, float* __restrict__ wmask,
                           int Bm, int n, int m, int k, float r2) {
  int gid = blockIdx.x * blockDim.x + threadIdx.x;
  if (gid >= Bm) return;
  int b = gid / m;
  const float* P = pos + (size_t)b * n * 3;
  float qx = qpos[(size_t)gid * 3 + 0];
  float qy = qpos[(size_t)gid * 3 + 1];
  float qz = qpos[(size_t)gid * 3 + 2];
  float bd[64]; int bi[64];
  float worstd = -1.f; int worst = 0;
  for (int p = 0; p < n; ++p) {
    float dx = P[p * 3 + 0] - qx;
    float dy = P[p * 3 + 1] - qy;
    float dz = P[p * 3 + 2] - qz;
    float d = dx * dx + dy * dy + dz * dz;
    if (p < k) {
      bd[p] = d; bi[p] = p;
      if (d > worstd) { worstd = d; worst = p; }
    } else if (d < worstd) {
      bd[worst] = d; bi[worst] = p;
      worstd = bd[0]; worst = 0;
      for (int j = 1; j < k; ++j)
        if (bd[j] > worstd) { worstd = bd[j]; worst = j; }
    }
  }
  size_t base = (size_t)gid * k;
  for (int j = 0; j < k; ++j) {
    nidx[base + j] = bi[j];
    if (wmask) wmask[base + j] = (bd[j] <= r2) ? 1.f : 0.f;
  }
}

// ======================= SA feature build: [x_j, pos_j - qpos, zero-pad] =======================

__global__ void build_feat_kernel(const float* __restrict__ x, const float* __restrict__ pos,
                                  const float* __restrict__ qpos, const int* __restrict__ nidx,
                                  float* __restrict__ feat,
                                  int n, int m, int k, int Cx, int ldF, long long total) {
  long long e = (long long)blockIdx.x * blockDim.x + threadIdx.x;
  if (e >= total) return;
  long long row = e / ldF; int c = (int)(e % ldF);
  long long bm = row / k;                 // b*m + q
  int b = (int)(bm / m);
  float v;
  if (c < Cx) {
    int j = nidx[row];
    v = x[((size_t)b * n + j) * Cx + c];
  } else if (c < Cx + 3) {
    int j = nidx[row];
    int d = c - Cx;
    v = pos[((size_t)b * n + j) * 3 + d] - qpos[(size_t)bm * 3 + d];
  } else {
    v = 0.f;
  }
  feat[e] = v;
}

// concat [a | b2 | zero-pad] with padded row stride
__global__ void concat2_pad_kernel(const float* __restrict__ a, const float* __restrict__ b2,
                                   float* __restrict__ dst, int C1, int C2, int ldF,
                                   long long total) {
  long long e = (long long)blockIdx.x * blockDim.x + threadIdx.x;
  if (e >= total) return;
  long long r = e / ldF; int c = (int)(e % ldF);
  float v;
  if (c < C1) v = a[(size_t)r * C1 + c];
  else if (c < C1 + C2) v = b2[(size_t)r * C2 + (c - C1)];
  else v = 0.f;
  dst[e] = v;
}

// ======================= WMMA GEMM (padded, branch-free inner loop) =======================
// A: RloadPad x CinPad (f32, zero-padded), Wt: CoutPad x CinPad (f16, transposed, padded).
// One wave computes a 16 x (16*NT) strip; A fragment reused across NT wmmas per K-chunk.

template <int NT>
__global__ void wmma_gemm_kernel(const float* __restrict__ A, const _Float16* __restrict__ Wt,
                                 const float* __restrict__ bias, float* __restrict__ out,
                                 int Rstore, int CinPad, int Cout, int ldOut) {
  int tileM  = blockIdx.x * 16;
  int tileN0 = blockIdx.y * (16 * NT);
  int lane = threadIdx.x;
  int half = lane >> 4;
  int l    = lane & 15;
  int arow = tileM + l;
  const float4* A4 = (const float4*)(A + (size_t)arow * CinPad);
  v8f acc[NT];
#pragma unroll
  for (int t = 0; t < NT; ++t) acc[t] = (v8f){0.f,0.f,0.f,0.f,0.f,0.f,0.f,0.f};

  for (int k0 = 0; k0 < CinPad; k0 += 32) {
    int a4 = (k0 >> 2) + half * 2;
    float4 fa = A4[a4 + 0];
    float4 fb = A4[a4 + 1];
    float4 fc = A4[a4 + 4];
    float4 fd = A4[a4 + 5];
    v16h a;
    a[0]  = (_Float16)fa.x; a[1]  = (_Float16)fa.y; a[2]  = (_Float16)fa.z; a[3]  = (_Float16)fa.w;
    a[4]  = (_Float16)fb.x; a[5]  = (_Float16)fb.y; a[6]  = (_Float16)fb.z; a[7]  = (_Float16)fb.w;
    a[8]  = (_Float16)fc.x; a[9]  = (_Float16)fc.y; a[10] = (_Float16)fc.z; a[11] = (_Float16)fc.w;
    a[12] = (_Float16)fd.x; a[13] = (_Float16)fd.y; a[14] = (_Float16)fd.z; a[15] = (_Float16)fd.w;
    int bbase = k0 + half * 8;
#pragma unroll
    for (int t = 0; t < NT; ++t) {
      const v8h* Wp = (const v8h*)(Wt + (size_t)(tileN0 + t * 16 + l) * CinPad + bbase);
      v8h b0 = Wp[0];
      v8h b1 = Wp[2];
      v16h bf;
#pragma unroll
      for (int j = 0; j < 8; ++j) { bf[j] = b0[j]; bf[8 + j] = b1[j]; }
      acc[t] = __builtin_amdgcn_wmma_f32_16x16x32_f16(false, a, false, bf,
                                                      (short)0, acc[t], false, false);
    }
  }
#pragma unroll
  for (int t = 0; t < NT; ++t) {
    int col = tileN0 + t * 16 + l;
    if (col < Cout) {
      float bv = bias ? bias[col] : 0.f;
#pragma unroll
      for (int r = 0; r < 8; ++r) {
        int row = tileM + r + half * 8;
        if (row < Rstore) out[(size_t)row * ldOut + col] = acc[t][r] + bv;
      }
    }
  }
}

// Fused variant: masked max-pool over contiguous groups of Kg rows (u32-encoded atomic max).
template <int NT>
__global__ void wmma_gemm_maxpool_kernel(const float* __restrict__ A, const _Float16* __restrict__ Wt,
                                         const float* __restrict__ bias,
                                         const float* __restrict__ wmask,
                                         unsigned* __restrict__ pool,
                                         int Rstore, int CinPad, int Cout, int Kg) {
  int tileM  = blockIdx.x * 16;
  int tileN0 = blockIdx.y * (16 * NT);
  int lane = threadIdx.x;
  int half = lane >> 4;
  int l    = lane & 15;
  int arow = tileM + l;
  const float4* A4 = (const float4*)(A + (size_t)arow * CinPad);
  v8f acc[NT];
#pragma unroll
  for (int t = 0; t < NT; ++t) acc[t] = (v8f){0.f,0.f,0.f,0.f,0.f,0.f,0.f,0.f};

  for (int k0 = 0; k0 < CinPad; k0 += 32) {
    int a4 = (k0 >> 2) + half * 2;
    float4 fa = A4[a4 + 0];
    float4 fb = A4[a4 + 1];
    float4 fc = A4[a4 + 4];
    float4 fd = A4[a4 + 5];
    v16h a;
    a[0]  = (_Float16)fa.x; a[1]  = (_Float16)fa.y; a[2]  = (_Float16)fa.z; a[3]  = (_Float16)fa.w;
    a[4]  = (_Float16)fb.x; a[5]  = (_Float16)fb.y; a[6]  = (_Float16)fb.z; a[7]  = (_Float16)fb.w;
    a[8]  = (_Float16)fc.x; a[9]  = (_Float16)fc.y; a[10] = (_Float16)fc.z; a[11] = (_Float16)fc.w;
    a[12] = (_Float16)fd.x; a[13] = (_Float16)fd.y; a[14] = (_Float16)fd.z; a[15] = (_Float16)fd.w;
    int bbase = k0 + half * 8;
#pragma unroll
    for (int t = 0; t < NT; ++t) {
      const v8h* Wp = (const v8h*)(Wt + (size_t)(tileN0 + t * 16 + l) * CinPad + bbase);
      v8h b0 = Wp[0];
      v8h b1 = Wp[2];
      v16h bf;
#pragma unroll
      for (int j = 0; j < 8; ++j) { bf[j] = b0[j]; bf[8 + j] = b1[j]; }
      acc[t] = __builtin_amdgcn_wmma_f32_16x16x32_f16(false, a, false, bf,
                                                      (short)0, acc[t], false, false);
    }
  }
#pragma unroll
  for (int t = 0; t < NT; ++t) {
    int col = tileN0 + t * 16 + l;
    if (col < Cout) {
      float bv = bias ? bias[col] : 0.f;
#pragma unroll
      for (int r = 0; r < 8; ++r) {
        int row = tileM + r + half * 8;
        if (row < Rstore) {
          float wv = wmask ? wmask[row] : 1.f;
          if (wv != 0.f) {
            float v = acc[t][r] + bv;
            unsigned u = __float_as_uint(v);
            u = (u & 0x80000000u) ? ~u : (u | 0x80000000u);
            atomicMax(&pool[(size_t)(row / Kg) * Cout + col], u);
          }
        }
      }
    }
  }
}

// ======================= masked BN: stats + apply =======================

__global__ void bn_stats_kernel(const float* __restrict__ y, const float* __restrict__ w,
                                float* __restrict__ S1, float* __restrict__ S2,
                                int R, int C, int rowsPerChunk) {
  __shared__ float sh1[TPB];
  __shared__ float sh2[TPB];
  int c = blockIdx.x;
  int r0 = blockIdx.y * rowsPerChunk;
  int r1 = r0 + rowsPerChunk; if (r1 > R) r1 = R;
  float s1 = 0.f, s2 = 0.f;
  for (int r = r0 + threadIdx.x; r < r1; r += blockDim.x) {
    float wv = w ? w[r] : 1.f;
    float v = y[(size_t)r * C + c];
    s1 += wv * v;
    s2 += wv * v * v;
  }
  sh1[threadIdx.x] = s1; sh2[threadIdx.x] = s2;
  __syncthreads();
  for (int k = TPB / 2; k > 0; k >>= 1) {
    if ((int)threadIdx.x < k) {
      sh1[threadIdx.x] += sh1[threadIdx.x + k];
      sh2[threadIdx.x] += sh2[threadIdx.x + k];
    }
    __syncthreads();
  }
  if (threadIdx.x == 0) {
    atomicAdd(&S1[c], sh1[0]);
    atomicAdd(&S2[c], sh2[0]);
  }
}

__global__ void bn_apply_kernel(float* __restrict__ y, const float* __restrict__ S1,
                                const float* __restrict__ S2, const float* __restrict__ nsum,
                                float ncst, const float* __restrict__ g,
                                const float* __restrict__ be, long long total, int C, int dorelu) {
  long long e = (long long)blockIdx.x * blockDim.x + threadIdx.x;
  if (e >= total) return;
  int c = (int)(e % C);
  float n = nsum ? nsum[0] : ncst;
  float mean = S1[c] / n;
  float var  = S2[c] / n - mean * mean;
  var = fmaxf(var, 0.f);
  float v = g[c] * (y[e] - mean) * rsqrtf(var + 1e-5f) + be[c];
  if (dorelu) v = fmaxf(v, 0.f);
  y[e] = v;
}

// ======================= TD pooling: max over gathered neighbors =======================

__global__ void gather_max_kernel(const float* __restrict__ h, const int* __restrict__ nidx,
                                  float* __restrict__ out, int n, int m, int k, int C,
                                  long long total) {
  long long e = (long long)blockIdx.x * blockDim.x + threadIdx.x;
  if (e >= total) return;
  long long r = e / C; int c = (int)(e % C);     // r = b*m + q
  int b = (int)(r / m);
  const int* NI = nidx + (size_t)r * k;
  float mx = -3.4e38f;
  for (int j = 0; j < k; ++j)
    mx = fmaxf(mx, h[((size_t)b * n + NI[j]) * C + c]);
  out[e] = mx;
}

__global__ void mean_rows_kernel(const float* __restrict__ h, float* __restrict__ out,
                                 int M, int C, long long total) {
  long long e = (long long)blockIdx.x * blockDim.x + threadIdx.x;
  if (e >= total) return;
  long long b = e / C; int c = (int)(e % C);
  float s = 0.f;
  for (int q = 0; q < M; ++q) s += h[((size_t)b * M + q) * C + c];
  out[e] = s / (float)M;
}

// ======================= host orchestration =======================

static inline dim3 eb(long long n) { return dim3((unsigned)((n + TPB - 1) / TPB)); }

extern "C" void kernel_launch(void* const* d_in, const int* in_sizes, int n_in,
                              void* d_out, int out_size, void* d_ws, size_t ws_size,
                              hipStream_t stream) {
  (void)in_sizes; (void)n_in; (void)out_size; (void)ws_size;
  const float* const* in = (const float* const*)d_in;

  const float* data  = in[0];
  const float *s1W1 = in[1], *s1b1 = in[2], *s1g1 = in[3], *s1e1 = in[4];
  const float *s1W2 = in[5], *s1b2 = in[6], *s1g2 = in[7], *s1e2 = in[8];
  const float *s1W3 = in[9], *s1b3 = in[10];
  const float *t1W = in[11], *t1b = in[12], *t1g = in[13], *t1e = in[14];
  const float *s2W1 = in[15], *s2b1 = in[16], *s2g1 = in[17], *s2e1 = in[18];
  const float *s2W2 = in[19], *s2b2 = in[20], *s2g2 = in[21], *s2e2 = in[22];
  const float *s2W3 = in[23], *s2b3 = in[24];
  const float *t2W = in[25], *t2b = in[26], *t2g = in[27], *t2e = in[28];
  const float *s3W1 = in[29], *s3b1 = in[30], *s3g1 = in[31], *s3e1 = in[32];
  const float *s3W2 = in[33], *s3b2 = in[34], *s3g2 = in[35], *s3e2 = in[36];
  const float *s3W3 = in[37], *s3b3 = in[38];
  const float *hW1 = in[39], *hb1 = in[40], *hW2 = in[41], *hb2 = in[42];
  const float *hW3 = in[43], *hb3 = in[44];

  const int Bb = 8, N = 4096;
  const int M1 = 2048, K1 = 64;  const long long R1 = (long long)Bb * M1 * K1;    // 1048576
  const int M2 = 512,  Ktd = 16; const long long Rt1 = (long long)Bb * M1;        // 16384
  const int M3 = 256,  K2 = 64;  const long long R2 = (long long)Bb * M3 * K2;    // 131072
  const int M4 = 64;             const long long Rt2 = (long long)Bb * M3;        // 2048
  const long long R3 = (long long)Bb * M4;                                        // 512

  // ---- workspace bump allocator ----
  size_t off = 0;
  char* base = (char*)d_ws;
  auto alloc = [&](size_t bytes) -> void* {
    void* p = base + off;
    off += (bytes + 255) & ~(size_t)255;
    return p;
  };
  float*    qpos1 = (float*)alloc((size_t)Bb * M1 * 3 * 4);
  float*    qpos2 = (float*)alloc((size_t)Bb * M2 * 3 * 4);
  float*    qpos3 = (float*)alloc((size_t)Bb * M3 * 3 * 4);
  float*    qpos4 = (float*)alloc((size_t)Bb * M4 * 3 * 4);
  int*      idxb  = (int*)alloc((size_t)Bb * M1 * 4);
  int*      nidx  = (int*)alloc((size_t)Bb * M1 * K1 * 4);
  float*    wmask = (float*)alloc((size_t)Bb * M1 * K1 * 4);
  float*    feat  = (float*)alloc((size_t)R2 * 288 * 4);       // max padded feat (SA2): 151 MB
  float*    hA    = (float*)alloc((size_t)R1 * 64 * 4);        // 256 MB
  float*    hB    = (float*)alloc((size_t)R1 * 64 * 4);        // 256 MB
  float*    x1    = (float*)alloc((size_t)Rt1 * 128 * 4);
  float*    x2    = (float*)alloc((size_t)Bb * M2 * 256 * 4);
  float*    x3    = (float*)alloc((size_t)Rt2 * 256 * 4);
  float*    x4    = (float*)alloc((size_t)Bb * M4 * 512 * 4);
  float*    gfeat = (float*)alloc((size_t)16 * 1024 * 4);      // 8 rows + 8 zero pad rows
  _Float16* wbuf  = (_Float16*)alloc((size_t)1024 * 1024 * 2); // padded transposed f16 weights
  float*    Sbuf  = (float*)alloc(2048 * 4);                   // S1 | S2
  float*    nsum  = (float*)alloc(256);
  float* S1 = Sbuf, *S2 = Sbuf + 1024;

  const unsigned NEG_INIT = 0x00800000u;  // enc(-FLT_MAX)

  auto prepW = [&](const float* W, int Cin, int Cout, int CinPad, int CoutPad) {
    long long tot = (long long)CoutPad * CinPad;
    prep_weight_kernel<<<eb(tot), TPB, 0, stream>>>(W, wbuf, Cin, Cout, CinPad, tot);
  };
  // Rload rows are covered by the grid (must be mult of 16 and fully allocated/zero-padded).
  auto gemm = [&](const float* A, const float* W, const float* bias, float* out,
                  long long Rload, long long Rstore, int Cin, int CinPad, int Cout, int ldOut) {
    int CoutPad = (Cout + 15) & ~15;
    prepW(W, Cin, Cout, CinPad, CoutPad);
    int ntiles = CoutPad / 16;
    if (ntiles % 4 == 0) {
      dim3 g((unsigned)(Rload / 16), (unsigned)(ntiles / 4));
      wmma_gemm_kernel<4><<<g, dim3(32), 0, stream>>>(A, wbuf, bias, out,
                                                      (int)Rstore, CinPad, Cout, ldOut);
    } else {
      dim3 g((unsigned)(Rload / 16), (unsigned)ntiles);
      wmma_gemm_kernel<1><<<g, dim3(32), 0, stream>>>(A, wbuf, bias, out,
                                                      (int)Rstore, CinPad, Cout, ldOut);
    }
  };
  auto gemm_max = [&](const float* A, const float* W, const float* bias, const float* wm,
                      unsigned* pool, long long R, int Cin, int CinPad, int Cout, int Kg) {
    int CoutPad = (Cout + 15) & ~15;
    prepW(W, Cin, Cout, CinPad, CoutPad);
    int ntiles = CoutPad / 16;
    dim3 g((unsigned)(R / 16), (unsigned)(ntiles / 4));
    wmma_gemm_maxpool_kernel<4><<<g, dim3(32), 0, stream>>>(A, wbuf, bias, wm, pool,
                                                            (int)R, CinPad, Cout, Kg);
  };
  auto bn = [&](float* y, const float* wm, const float* np, float ncst,
                const float* gam, const float* bet, long long R, int C, int dorelu) {
    fill_f32_kernel<<<eb(2048), TPB, 0, stream>>>(Sbuf, 0.f, 2048);
    int chunks = (int)((R + 8191) / 8192); if (chunks > 64) chunks = 64; if (chunks < 1) chunks = 1;
    int rpc = (int)((R + chunks - 1) / chunks);
    bn_stats_kernel<<<dim3((unsigned)C, (unsigned)chunks), TPB, 0, stream>>>(
        y, wm, S1, S2, (int)R, C, rpc);
    long long tot = R * C;
    bn_apply_kernel<<<eb(tot), TPB, 0, stream>>>(y, S1, S2, np, ncst, gam, bet, tot, C, dorelu);
  };
  auto masksum = [&](const float* wm, long long R) {
    fill_f32_kernel<<<1, 1, 0, stream>>>(nsum, 0.f, 1);
    unsigned nb = (unsigned)((R + TPB - 1) / TPB); if (nb > 1024) nb = 1024;
    reduce_sum_kernel<<<nb, TPB, 0, stream>>>(wm, R, nsum);
  };

  // ================= SA1 =================
  fps_kernel<<<Bb, TPB, 0, stream>>>(data, idxb, N, M1);
  gather_rows_kernel<<<eb((long long)Bb * M1 * 3), TPB, 0, stream>>>(
      data, idxb, qpos1, N, M1, 3, (long long)Bb * M1 * 3);
  knn_kernel<<<eb((long long)Bb * M1), TPB, 0, stream>>>(
      data, qpos1, nidx, wmask, Bb * M1, N, M1, K1, 0.2f * 0.2f);
  build_feat_kernel<<<eb(R1 * 32), TPB, 0, stream>>>(
      data, data, qpos1, nidx, feat, N, M1, K1, 3, 32, R1 * 32);
  masksum(wmask, R1);
  gemm(feat, s1W1, s1b1, hA, R1, R1, 6, 32, 64, 64);
  bn(hA, wmask, nsum, 0.f, s1g1, s1e1, R1, 64, 1);
  gemm(hA, s1W2, s1b2, hB, R1, R1, 64, 64, 64, 64);
  bn(hB, wmask, nsum, 0.f, s1g2, s1e2, R1, 64, 1);
  fill_u32_kernel<<<eb(Rt1 * 128), TPB, 0, stream>>>((unsigned*)x1, NEG_INIT, Rt1 * 128);
  gemm_max(hB, s1W3, s1b3, wmask, (unsigned*)x1, R1, 64, 64, 128, K1);
  decode_max_kernel<<<eb(Rt1 * 128), TPB, 0, stream>>>((unsigned*)x1, Rt1 * 128);

  // ================= TD1 =================
  fps_kernel<<<Bb, TPB, 0, stream>>>(qpos1, idxb, M1, M2);
  gather_rows_kernel<<<eb((long long)Bb * M2 * 3), TPB, 0, stream>>>(
      qpos1, idxb, qpos2, M1, M2, 3, (long long)Bb * M2 * 3);
  knn_kernel<<<eb((long long)Bb * M2), TPB, 0, stream>>>(
      qpos1, qpos2, nidx, nullptr, Bb * M2, M1, M2, Ktd, 0.f);
  gemm(x1, t1W, t1b, hA, Rt1, Rt1, 128, 128, 256, 256);
  bn(hA, nullptr, nullptr, (float)Rt1, t1g, t1e, Rt1, 256, 1);
  gather_max_kernel<<<eb((long long)Bb * M2 * 256), TPB, 0, stream>>>(
      hA, nidx, x2, M1, M2, Ktd, 256, (long long)Bb * M2 * 256);

  // ================= SA2 =================
  fps_kernel<<<Bb, TPB, 0, stream>>>(qpos2, idxb, M2, M3);
  gather_rows_kernel<<<eb((long long)Bb * M3 * 3), TPB, 0, stream>>>(
      qpos2, idxb, qpos3, M2, M3, 3, (long long)Bb * M3 * 3);
  knn_kernel<<<eb((long long)Bb * M3), TPB, 0, stream>>>(
      qpos2, qpos3, nidx, wmask, Bb * M3, M2, M3, K2, 0.4f * 0.4f);
  build_feat_kernel<<<eb(R2 * 288), TPB, 0, stream>>>(
      x2, qpos2, qpos3, nidx, feat, M2, M3, K2, 256, 288, R2 * 288);
  masksum(wmask, R2);
  gemm(feat, s2W1, s2b1, hA, R2, R2, 259, 288, 128, 128);
  bn(hA, wmask, nsum, 0.f, s2g1, s2e1, R2, 128, 1);
  gemm(hA, s2W2, s2b2, hB, R2, R2, 128, 128, 128, 128);
  bn(hB, wmask, nsum, 0.f, s2g2, s2e2, R2, 128, 1);
  fill_u32_kernel<<<eb(Rt2 * 256), TPB, 0, stream>>>((unsigned*)x3, NEG_INIT, Rt2 * 256);
  gemm_max(hB, s2W3, s2b3, wmask, (unsigned*)x3, R2, 128, 128, 256, K2);
  decode_max_kernel<<<eb(Rt2 * 256), TPB, 0, stream>>>((unsigned*)x3, Rt2 * 256);

  // ================= TD2 =================
  fps_kernel<<<Bb, TPB, 0, stream>>>(qpos3, idxb, M3, M4);
  gather_rows_kernel<<<eb((long long)Bb * M4 * 3), TPB, 0, stream>>>(
      qpos3, idxb, qpos4, M3, M4, 3, (long long)Bb * M4 * 3);
  knn_kernel<<<eb((long long)Bb * M4), TPB, 0, stream>>>(
      qpos3, qpos4, nidx, nullptr, Bb * M4, M3, M4, Ktd, 0.f);
  gemm(x3, t2W, t2b, hA, Rt2, Rt2, 256, 256, 512, 512);
  bn(hA, nullptr, nullptr, (float)Rt2, t2g, t2e, Rt2, 512, 1);
  gather_max_kernel<<<eb((long long)Bb * M4 * 512), TPB, 0, stream>>>(
      hA, nidx, x4, M3, M4, Ktd, 512, (long long)Bb * M4 * 512);

  // ================= SA3 (group-all MLP) + mean =================
  concat2_pad_kernel<<<eb(R3 * 544), TPB, 0, stream>>>(x4, qpos4, feat, 512, 3, 544, R3 * 544);
  gemm(feat, s3W1, s3b1, hA, R3, R3, 515, 544, 256, 256);
  bn(hA, nullptr, nullptr, (float)R3, s3g1, s3e1, R3, 256, 1);
  gemm(hA, s3W2, s3b2, hB, R3, R3, 256, 256, 512, 512);
  bn(hB, nullptr, nullptr, (float)R3, s3g2, s3e2, R3, 512, 1);
  gemm(hB, s3W3, s3b3, hA, R3, R3, 512, 512, 1024, 1024);
  mean_rows_kernel<<<eb((long long)Bb * 1024), TPB, 0, stream>>>(
      hA, gfeat, M4, 1024, (long long)Bb * 1024);
  // zero the 8 pad rows of gfeat so head GEMM loads are well-defined
  fill_f32_kernel<<<eb(8 * 1024), TPB, 0, stream>>>(gfeat + 8 * 1024, 0.f, 8 * 1024);

  // ================= head (rows padded 8 -> 16; pad rows produce bias, harmless) ======
  gemm(gfeat, hW1, hb1, hA, 16, 16, 1024, 1024, 512, 512);
  relu_kernel<<<eb(16 * 512), TPB, 0, stream>>>(hA, 16 * 512);
  gemm(hA, hW2, hb2, hB, 16, 16, 512, 512, 256, 256);
  relu_kernel<<<eb(16 * 256), TPB, 0, stream>>>(hB, 16 * 256);
  gemm(hB, hW3, hb3, (float*)d_out, 16, Bb, 256, 256, 13, 13);
}
